// LanguageModel_74577812128349
// MI455X (gfx1250) — matrix operand order repaired and checked by reading, not compiled, not run on hardware
//
#include <hip/hip_runtime.h>
#include <hip/hip_bf16.h>

typedef _Float16 f16;
typedef __attribute__((ext_vector_type(16))) _Float16 v16h;
typedef __attribute__((ext_vector_type(8)))  _Float16 v8h;
typedef __attribute__((ext_vector_type(8)))  float    v8f;
typedef int i32x4 __attribute__((vector_size(16)));   // matches builtin's V4i

#define D_MODEL 1024
#define N_HEADS 16
#define D_HEAD  64
#define N_LAYERS 12
#define VOCAB   32000
#define BATCH   8
#define SEQ     512
#define D_FF    4096
#define TAU_F   30.0f
#define NEG_INF -1e9f

#define LDS_STRIDE 40  // 32 f16 payload + 8 pad (80B rows; keeps 16B alignment)

static __device__ __forceinline__ v16h cat16(v8h lo, v8h hi) {
  v16h r;
#pragma unroll
  for (int i = 0; i < 8; ++i) { r[i] = lo[i]; r[i + 8] = hi[i]; }
  return r;
}

// 16-byte global -> LDS copy. Prefers the CDNA5 async LDS-DMA path
// (GLOBAL_LOAD_ASYNC_TO_LDS_B128, tracked by ASYNCcnt); falls back to a
// synchronous load+ds_store if the builtin is unavailable.
static __device__ __forceinline__ void copy16_g2l(const f16* g, f16* l) {
#if defined(__has_builtin) && __has_builtin(__builtin_amdgcn_global_load_async_to_lds_b128)
  __builtin_amdgcn_global_load_async_to_lds_b128(
      (__attribute__((address_space(1))) i32x4*)(void*)(f16*)g,
      (__attribute__((address_space(3))) i32x4*)(void*)l, 0, 0);
#else
  *(v8h*)l = *(const v8h*)g;
#endif
}

static __device__ __forceinline__ void wait_g2l() {
#if defined(__has_builtin) && __has_builtin(__builtin_amdgcn_s_wait_asynccnt)
  __builtin_amdgcn_s_wait_asynccnt(0);
#endif
}

// ---------------------------------------------------------------------------
// Generic batched NT GEMM on WMMA: C[m,n] = sum_k A[m,k] * B[n,k]
// A: f16 [M,K] row-major (lda), B: f16 [N,K] row-major (ldb).
// Output either fp32 (Cf) or f16 (Ch); exactly one is non-null.
// blockIdx.z = bb*nh + hh with per-(bb,hh) element offsets for A/B/C.
// 128-thread block (4 waves) computes a 64x64 C tile. A/B 64x32 K-tiles are
// double-buffered in LDS via async global->LDS copies; each wave does 2x2
// v_wmma_f32_16x16x32_f16 fragments per K-step read from LDS.
// ---------------------------------------------------------------------------
__global__ void __launch_bounds__(128) gemm_nt_wmma(
    const f16* __restrict__ A, const f16* __restrict__ B,
    float* __restrict__ Cf, f16* __restrict__ Ch,
    int M, int N, int K, int lda, int ldb, int ldc, int nh,
    long long offA_b, long long offA_h,
    long long offB_b, long long offB_h,
    long long offC_b, long long offC_h)
{
  __shared__ alignas(16) f16 sA[2][64 * LDS_STRIDE];
  __shared__ alignas(16) f16 sB[2][64 * LDS_STRIDE];

  const int z  = blockIdx.z;
  const int bb = z / nh;
  const int hh = z - bb * nh;
  A += (long long)bb * offA_b + (long long)hh * offA_h;
  B += (long long)bb * offB_b + (long long)hh * offB_h;
  const long long coff = (long long)bb * offC_b + (long long)hh * offC_h;

  const int lane = threadIdx.x & 31;
  const int wave = threadIdx.x >> 5;
  const int wm = wave >> 1;          // 0..1 : M half of block tile
  const int wn = wave & 1;           // 0..1 : N half of block tile
  const int m0blk = blockIdx.y * 64;
  const int n0blk = blockIdx.x * 64;

  const int fr = lane & 15;          // A row / B column within a 16-tile
  const int kh = lane >> 4;          // K-half selector

  // Stage one 64x32 K-tile of A and B into LDS buffer `buf`.
  // 64 rows x 4 16B-chunks = 256 chunks per matrix; 128 threads x 2 chunks.
  auto stage = [&](int kt, int buf) {
    const long long k0 = (long long)kt * 32;
#pragma unroll
    for (int j = 0; j < 2; ++j) {
      const int c   = threadIdx.x + j * 128;
      const int row = c >> 2;
      const int sub = (c & 3) * 8;
      copy16_g2l(A + (long long)(m0blk + row) * lda + k0 + sub,
                 &sA[buf][row * LDS_STRIDE + sub]);
      copy16_g2l(B + (long long)(n0blk + row) * ldb + k0 + sub,
                 &sB[buf][row * LDS_STRIDE + sub]);
    }
  };

  const int nk = K >> 5;             // K / 32 (all K are multiples of 32)

  v8f c00 = {}, c01 = {}, c10 = {}, c11 = {};

  stage(0, 0);
  for (int kt = 0; kt < nk; ++kt) {
    const int buf = kt & 1;
    wait_g2l();                      // this wave's copies for tile kt done
    __syncthreads();                 // -> everyone's copies done, computes of
                                     //    tile kt-1 finished on all waves
    if (kt + 1 < nk) stage(kt + 1, buf ^ 1);

    // A fragment (ISA 16-bit A 16x32 layout): lane holds row fr,
    // K = kh*8..+7 (VGPR0-3) and K = 16+kh*8..+7 (VGPR4-7).
    const f16* pa = &sA[buf][(wm * 32 + fr) * LDS_STRIDE + kh * 8];
    v16h a0 = cat16(*(const v8h*)pa, *(const v8h*)(pa + 16));
    v16h a1 = cat16(*(const v8h*)(pa + 16 * LDS_STRIDE),
                    *(const v8h*)(pa + 16 * LDS_STRIDE + 16));
    // B fragment: lane holds column fr, K = kh*8..+7 and K = 16+kh*8..+7.
    const f16* pb = &sB[buf][(wn * 32 + fr) * LDS_STRIDE + kh * 8];
    v16h b0 = cat16(*(const v8h*)pb, *(const v8h*)(pb + 16));
    v16h b1 = cat16(*(const v8h*)(pb + 16 * LDS_STRIDE),
                    *(const v8h*)(pb + 16 * LDS_STRIDE + 16));

    c00 = __builtin_amdgcn_wmma_f32_16x16x32_f16(false, a0, false, b0, (short)0, c00, false, false);
    c01 = __builtin_amdgcn_wmma_f32_16x16x32_f16(false, a0, false, b1, (short)0, c01, false, false);
    c10 = __builtin_amdgcn_wmma_f32_16x16x32_f16(false, a1, false, b0, (short)0, c10, false, false);
    c11 = __builtin_amdgcn_wmma_f32_16x16x32_f16(false, a1, false, b1, (short)0, c11, false, false);
  }

  // C/D layout: lane -> n = lane%16; VGPR r -> m = (lane/16)*8 + r.
  const int cn = n0blk + wn * 32 + fr;
  const int cm = m0blk + wm * 32 + kh * 8;
  if (Ch) {
    f16* C = Ch + coff;
#pragma unroll
    for (int r = 0; r < 8; ++r) {
      const long long r0 = (long long)(cm + r) * ldc;
      const long long r1 = (long long)(cm + 16 + r) * ldc;
      C[r0 + cn]      = (f16)c00[r];
      C[r0 + cn + 16] = (f16)c01[r];
      C[r1 + cn]      = (f16)c10[r];
      C[r1 + cn + 16] = (f16)c11[r];
    }
  } else {
    float* C = Cf + coff;
#pragma unroll
    for (int r = 0; r < 8; ++r) {
      const long long r0 = (long long)(cm + r) * ldc;
      const long long r1 = (long long)(cm + 16 + r) * ldc;
      C[r0 + cn]      = c00[r];
      C[r0 + cn + 16] = c01[r];
      C[r1 + cn]      = c10[r];
      C[r1 + cn + 16] = c11[r];
    }
  }
}

// ---------------------------------------------------------------------------
// Elementwise / reduction kernels (fp32 math, f16 outputs where consumed by
// WMMA GEMMs)
// ---------------------------------------------------------------------------

__global__ void __launch_bounds__(256) embed_kernel(
    const int* __restrict__ ids, const float* __restrict__ Wv,
    float* __restrict__ x)
{
  long long i = (long long)blockIdx.x * blockDim.x + threadIdx.x;
  if (i >= (long long)BATCH * SEQ * D_MODEL) return;
  const int d = (int)(i & (D_MODEL - 1));
  const long long t = i >> 10;           // token index b*SEQ + l
  const int l = (int)(t & (SEQ - 1));
  const int id = ids[t];
  const float freq = expf(-logf(10000.0f) * (float)(d & ~1) / (float)D_MODEL);
  const float ang = (float)l * freq;
  const float pe = (d & 1) ? cosf(ang) : sinf(ang);
  x[i] = Wv[(long long)d * VOCAB + id] + pe;
}

__global__ void __launch_bounds__(256) layernorm_f16_kernel(
    const float* __restrict__ x, const float* __restrict__ gamma,
    const float* __restrict__ beta, f16* __restrict__ out)
{
  __shared__ float red[256];
  const int row = blockIdx.x;
  const float* xr = x + (long long)row * D_MODEL;
  float v[4];
  float s = 0.f;
#pragma unroll
  for (int i = 0; i < 4; ++i) { v[i] = xr[threadIdx.x + i * 256]; s += v[i]; }
  red[threadIdx.x] = s;
  __syncthreads();
  for (int off = 128; off > 0; off >>= 1) {
    if (threadIdx.x < off) red[threadIdx.x] += red[threadIdx.x + off];
    __syncthreads();
  }
  const float mean = red[0] * (1.0f / D_MODEL);
  __syncthreads();
  s = 0.f;
#pragma unroll
  for (int i = 0; i < 4; ++i) { const float d = v[i] - mean; s += d * d; }
  red[threadIdx.x] = s;
  __syncthreads();
  for (int off = 128; off > 0; off >>= 1) {
    if (threadIdx.x < off) red[threadIdx.x] += red[threadIdx.x + off];
    __syncthreads();
  }
  const float rstd = rsqrtf(red[0] * (1.0f / D_MODEL) + 1e-5f);
  f16* orow = out + (long long)row * D_MODEL;
#pragma unroll
  for (int i = 0; i < 4; ++i) {
    const int c = threadIdx.x + i * 256;
    orow[c] = (f16)(((v[i] - mean) * rstd) * gamma[c] + beta[c]);
  }
}

// tanh-clipped, causally+pad masked softmax over rows of S [B,H,L,L] -> f16
__global__ void __launch_bounds__(256) softmax_kernel(
    const float* __restrict__ S, const int* __restrict__ amask,
    f16* __restrict__ A)
{
  __shared__ float red[256];
  const int idx = blockIdx.x;            // (b*H + h)*L + l
  const int l  = idx & (SEQ - 1);
  const int bh = idx >> 9;
  const int b  = bh >> 4;
  const float* srow = S + (long long)idx * SEQ;
  const float scale = 0.125f;            // 1/sqrt(D_HEAD)
  float v[2];
  float mx = -3.4e38f;
#pragma unroll
  for (int i = 0; i < 2; ++i) {
    const int m = threadIdx.x + i * 256;
    float s = TAU_F * tanhf(srow[m] * scale);
    const bool masked = (m > l) || (amask[b * SEQ + m] == 0);
    v[i] = masked ? NEG_INF : s;
    mx = fmaxf(mx, v[i]);
  }
  red[threadIdx.x] = mx;
  __syncthreads();
  for (int off = 128; off > 0; off >>= 1) {
    if (threadIdx.x < off) red[threadIdx.x] = fmaxf(red[threadIdx.x], red[threadIdx.x + off]);
    __syncthreads();
  }
  mx = red[0];
  __syncthreads();
  float sum = 0.f;
#pragma unroll
  for (int i = 0; i < 2; ++i) { v[i] = expf(v[i] - mx); sum += v[i]; }
  red[threadIdx.x] = sum;
  __syncthreads();
  for (int off = 128; off > 0; off >>= 1) {
    if (threadIdx.x < off) red[threadIdx.x] += red[threadIdx.x + off];
    __syncthreads();
  }
  const float inv = 1.0f / red[0];
  f16* arow = A + (long long)idx * SEQ;
#pragma unroll
  for (int i = 0; i < 2; ++i) {
    const int m = threadIdx.x + i * 256;
    arow[m] = (f16)(v[i] * inv);
  }
}

// v [B*L, H*64] f16 -> vT [B,H,64,L] f16
__global__ void __launch_bounds__(256) transpose_v_kernel(
    const f16* __restrict__ v, f16* __restrict__ vT)
{
  long long i = (long long)blockIdx.x * blockDim.x + threadIdx.x;
  if (i >= (long long)BATCH * N_HEADS * D_HEAD * SEQ) return;
  const int l = (int)(i & (SEQ - 1));
  long long t = i >> 9;
  const int d = (int)(t & (D_HEAD - 1)); t >>= 6;
  const int h = (int)(t & (N_HEADS - 1));
  const int b = (int)(t >> 4);
  vT[i] = v[((long long)(b * SEQ + l)) * D_MODEL + h * D_HEAD + d];
}

__global__ void __launch_bounds__(256) f32_to_f16_kernel(
    const float* __restrict__ in, f16* __restrict__ out, long long n)
{
  long long i = (long long)blockIdx.x * blockDim.x + threadIdx.x;
  if (i < n) out[i] = (f16)in[i];
}

// out[layer][c][r] = (f16) in[layer][r][c]   (in: [nl, R, C] fp32)
__global__ void __launch_bounds__(256) transpose_f16_kernel(
    const float* __restrict__ in, f16* __restrict__ out, int nl, int R, int C)
{
  long long i = (long long)blockIdx.x * blockDim.x + threadIdx.x;
  const long long per = (long long)R * C;
  if (i >= (long long)nl * per) return;
  const int lyr = (int)(i / per);
  const long long rem = i - (long long)lyr * per;
  const int c = (int)(rem / R);
  const int r = (int)(rem - (long long)c * R);
  out[i] = (f16)in[(long long)lyr * per + (long long)r * C + c];
}

__global__ void __launch_bounds__(256) add2_kernel(
    const float* __restrict__ a, const float* __restrict__ b,
    float* __restrict__ o, long long n)
{
  long long i = (long long)blockIdx.x * blockDim.x + threadIdx.x;
  if (i < n) o[i] = a[i] + b[i];
}

__global__ void __launch_bounds__(256) bias_gelu_kernel(
    const float* __restrict__ t, const float* __restrict__ bias,
    f16* __restrict__ h, long long n, int ncols)
{
  long long i = (long long)blockIdx.x * blockDim.x + threadIdx.x;
  if (i >= n) return;
  const int c = (int)(i % ncols);
  const float x = t[i] + bias[c];
  h[i] = (f16)(0.5f * x * (1.0f + erff(x * 0.70710678118654752f)));
}

__global__ void __launch_bounds__(256) add_bias_res_kernel(
    const float* __restrict__ z1, const float* __restrict__ dn,
    const float* __restrict__ bias, float* __restrict__ x,
    long long n, int ncols)
{
  long long i = (long long)blockIdx.x * blockDim.x + threadIdx.x;
  if (i >= n) return;
  const int c = (int)(i % ncols);
  x[i] = z1[i] + dn[i] + bias[c];
}

// ---------------------------------------------------------------------------
// Host-side launch
// ---------------------------------------------------------------------------
extern "C" void kernel_launch(void* const* d_in, const int* in_sizes, int n_in,
                              void* d_out, int out_size, void* d_ws, size_t ws_size,
                              hipStream_t stream) {
  (void)in_sizes; (void)n_in; (void)out_size; (void)ws_size;

  const int*   ids      = (const int*)d_in[0];
  const int*   amask    = (const int*)d_in[1];
  const float* W_vocab  = (const float*)d_in[2];
  const float* W_devoc  = (const float*)d_in[3];
  const float* WQ       = (const float*)d_in[4];
  const float* WK       = (const float*)d_in[5];
  const float* WV       = (const float*)d_in[6];
  const float* WO       = (const float*)d_in[7];
  const float* gamma1   = (const float*)d_in[8];
  const float* beta1    = (const float*)d_in[9];
  const float* gamma2   = (const float*)d_in[10];
  const float* beta2    = (const float*)d_in[11];
  const float* W_up     = (const float*)d_in[12];
  const float* b_up     = (const float*)d_in[13];
  const float* W_down   = (const float*)d_in[14];
  const float* b_down   = (const float*)d_in[15];
  const float* gamma_f  = (const float*)d_in[16];
  const float* beta_f   = (const float*)d_in[17];
  float* out = (float*)d_out;
  char*  ws  = (char*)d_ws;

  const size_t SZ_QKV = (size_t)N_LAYERS * D_MODEL * D_MODEL * sizeof(f16);
  const size_t SZ_FF  = (size_t)N_LAYERS * D_MODEL * D_FF * sizeof(f16);
  const size_t SZ_DV  = (size_t)VOCAB * D_MODEL * sizeof(f16);
  const size_t SZ_XF  = (size_t)BATCH * SEQ * D_MODEL * sizeof(float);
  const size_t SZ_XH  = SZ_XF / 2;
  const size_t SZ_S   = (size_t)BATCH * N_HEADS * SEQ * SEQ * sizeof(float);

  const size_t OFF_WQ  = 0;
  const size_t OFF_WK  = OFF_WQ + SZ_QKV;
  const size_t OFF_WV  = OFF_WK + SZ_QKV;
  const size_t OFF_WO  = OFF_WV + SZ_QKV;
  const size_t OFF_WUP = OFF_WO + SZ_QKV;
  const size_t OFF_WDN = OFF_WUP + SZ_FF;
  const size_t OFF_WDV = OFF_WDN + SZ_FF;
  const size_t OFF_X   = OFF_WDV + SZ_DV;
  const size_t OFF_Z1  = OFF_X + SZ_XF;
  const size_t OFF_O   = OFF_Z1 + SZ_XF;
  const size_t OFF_XN  = OFF_O + SZ_XF;
  const size_t OFF_Q   = OFF_XN + SZ_XH;
  const size_t OFF_KK  = OFF_Q + SZ_XH;
  const size_t OFF_V   = OFF_KK + SZ_XH;
  const size_t OFF_VT  = OFF_V + SZ_XH;
  const size_t OFF_ATT = OFF_VT + SZ_XH;
  const size_t OFF_S   = OFF_ATT + SZ_XH;  // fp32 scores; also FFN-up fp32 scratch
  const size_t OFF_A   = OFF_S + SZ_S;     // f16 probs; also FFN hidden f16

  f16*   WQh   = (f16*)(ws + OFF_WQ);
  f16*   WKh   = (f16*)(ws + OFF_WK);
  f16*   WVh   = (f16*)(ws + OFF_WV);
  f16*   WOh   = (f16*)(ws + OFF_WO);
  f16*   WupT  = (f16*)(ws + OFF_WUP);
  f16*   WdnT  = (f16*)(ws + OFF_WDN);
  f16*   WdvT  = (f16*)(ws + OFF_WDV);
  float* x     = (float*)(ws + OFF_X);
  float* z1    = (float*)(ws + OFF_Z1);
  float* o_tmp = (float*)(ws + OFF_O);
  f16*   xn    = (f16*)(ws + OFF_XN);
  f16*   q     = (f16*)(ws + OFF_Q);
  f16*   k     = (f16*)(ws + OFF_KK);
  f16*   v     = (f16*)(ws + OFF_V);
  f16*   vT    = (f16*)(ws + OFF_VT);
  f16*   attn  = (f16*)(ws + OFF_ATT);
  float* S     = (float*)(ws + OFF_S);
  float* t_ff  = (float*)(ws + OFF_S);
  f16*   Asoft = (f16*)(ws + OFF_A);
  f16*   hbuf  = (f16*)(ws + OFF_A);

  auto blocks = [](long long n) { return (unsigned)((n + 255) / 256); };

  auto gemm = [&](const f16* A, const f16* B, float* Cf, f16* Ch,
                  int M, int N, int K, int lda, int ldb, int ldc,
                  int nb, int nh,
                  long long oAb, long long oAh, long long oBb, long long oBh,
                  long long oCb, long long oCh) {
    dim3 grid((unsigned)(N / 64), (unsigned)(M / 64), (unsigned)(nb * nh));
    gemm_nt_wmma<<<grid, dim3(128), 0, stream>>>(A, B, Cf, Ch, M, N, K,
                                                 lda, ldb, ldc, nh,
                                                 oAb, oAh, oBb, oBh, oCb, oCh);
  };

  // ---- weight prep (deterministic, every call) ----
  {
    const long long nqkv = (long long)N_LAYERS * D_MODEL * D_MODEL;
    f32_to_f16_kernel<<<blocks(nqkv), 256, 0, stream>>>(WQ, WQh, nqkv);
    f32_to_f16_kernel<<<blocks(nqkv), 256, 0, stream>>>(WK, WKh, nqkv);
    f32_to_f16_kernel<<<blocks(nqkv), 256, 0, stream>>>(WV, WVh, nqkv);
    f32_to_f16_kernel<<<blocks(nqkv), 256, 0, stream>>>(WO, WOh, nqkv);
    const long long nff = (long long)N_LAYERS * D_MODEL * D_FF;
    transpose_f16_kernel<<<blocks(nff), 256, 0, stream>>>(W_up, WupT, N_LAYERS, D_MODEL, D_FF);
    transpose_f16_kernel<<<blocks(nff), 256, 0, stream>>>(W_down, WdnT, N_LAYERS, D_FF, D_MODEL);
    const long long ndv = (long long)D_MODEL * VOCAB;
    transpose_f16_kernel<<<blocks(ndv), 256, 0, stream>>>(W_devoc, WdvT, 1, D_MODEL, VOCAB);
  }

  const int M = BATCH * SEQ;  // 4096
  const long long nMD = (long long)M * D_MODEL;

  embed_kernel<<<blocks(nMD), 256, 0, stream>>>(ids, W_vocab, x);

  for (int i = 0; i < N_LAYERS; ++i) {
    const long long wqo = (long long)i * D_MODEL * D_MODEL;
    const long long wfo = (long long)i * D_MODEL * D_FF;

    // u = LN(x)
    layernorm_f16_kernel<<<M, 256, 0, stream>>>(x, gamma1 + i * D_MODEL, beta1 + i * D_MODEL, xn);

    // q/k/v = u @ W{Q,K,V}^T  (NT: weight stored [HD, D])
    gemm(xn, WQh + wqo, nullptr, q, M, D_MODEL, D_MODEL, D_MODEL, D_MODEL, D_MODEL,
         1, 1, 0, 0, 0, 0, 0, 0);
    gemm(xn, WKh + wqo, nullptr, k, M, D_MODEL, D_MODEL, D_MODEL, D_MODEL, D_MODEL,
         1, 1, 0, 0, 0, 0, 0, 0);
    gemm(xn, WVh + wqo, nullptr, v, M, D_MODEL, D_MODEL, D_MODEL, D_MODEL, D_MODEL,
         1, 1, 0, 0, 0, 0, 0, 0);

    transpose_v_kernel<<<blocks((long long)BATCH * N_HEADS * D_HEAD * SEQ), 256, 0, stream>>>(v, vT);

    // S[b,h] = q_h @ k_h^T   (batched NT, K = 64)
    gemm(q, k, S, nullptr, SEQ, SEQ, D_HEAD, D_MODEL, D_MODEL, SEQ,
         BATCH, N_HEADS,
         (long long)SEQ * D_MODEL, D_HEAD,
         (long long)SEQ * D_MODEL, D_HEAD,
         (long long)N_HEADS * SEQ * SEQ, (long long)SEQ * SEQ);

    softmax_kernel<<<BATCH * N_HEADS * SEQ, 256, 0, stream>>>(S, amask, Asoft);

    // attn[b,h] = A_h @ v_h   (NT against vT [64, L])
    gemm(Asoft, vT, nullptr, attn, SEQ, D_HEAD, SEQ, SEQ, SEQ, D_MODEL,
         BATCH, N_HEADS,
         (long long)N_HEADS * SEQ * SEQ, (long long)SEQ * SEQ,
         (long long)N_HEADS * D_HEAD * SEQ, (long long)D_HEAD * SEQ,
         (long long)SEQ * D_MODEL, D_HEAD);

    // o = attn @ W_O^T ; z1 = x + o
    gemm(attn, WOh + wqo, o_tmp, nullptr, M, D_MODEL, D_MODEL, D_MODEL, D_MODEL, D_MODEL,
         1, 1, 0, 0, 0, 0, 0, 0);
    add2_kernel<<<blocks(nMD), 256, 0, stream>>>(x, o_tmp, z1, nMD);

    // v1 = LN(z1)
    layernorm_f16_kernel<<<M, 256, 0, stream>>>(z1, gamma2 + i * D_MODEL, beta2 + i * D_MODEL, xn);

    // h = gelu(v1 @ W_up + b_up)
    gemm(xn, WupT + wfo, t_ff, nullptr, M, D_FF, D_MODEL, D_MODEL, D_MODEL, D_FF,
         1, 1, 0, 0, 0, 0, 0, 0);
    bias_gelu_kernel<<<blocks((long long)M * D_FF), 256, 0, stream>>>(
        t_ff, b_up + i * D_FF, hbuf, (long long)M * D_FF, D_FF);

    // x = z1 + h @ W_down + b_down
    gemm(hbuf, WdnT + wfo, o_tmp, nullptr, M, D_MODEL, D_FF, D_FF, D_FF, D_MODEL,
         1, 1, 0, 0, 0, 0, 0, 0);
    add_bias_res_kernel<<<blocks(nMD), 256, 0, stream>>>(
        z1, o_tmp, b_down + i * D_MODEL, x, nMD, D_MODEL);
  }

  // final LN + devocab projection (fp32 logits straight into d_out)
  layernorm_f16_kernel<<<M, 256, 0, stream>>>(x, gamma_f, beta_f, xn);
  gemm(xn, WdvT, out, nullptr, M, VOCAB, D_MODEL, D_MODEL, D_MODEL, VOCAB,
       1, 1, 0, 0, 0, 0, 0, 0);
}
